// InterpNet_18588618457477
// MI455X (gfx1250) — matrix operand order repaired and checked by reading, not compiled, not run on hardware
//
#include <hip/hip_runtime.h>
#include <hip/hip_bf16.h>
#include <math.h>

typedef __attribute__((ext_vector_type(16))) _Float16 v16h;
typedef __attribute__((ext_vector_type(8)))  _Float16 v8h;
typedef __attribute__((ext_vector_type(8)))  float    v8f;

#define NSRC   16384
#define NTGT   16384
#define KNN    16
#define LDIM   128
#define ETOT   (NTGT * KNN)       // 262144 edges
#define NTILES (ETOT / 16)        // 16384 wave-tiles
#define STRIDE 136                // f16 LDS row stride (272B = 17*16B, keeps 16B alignment)
#define LL     (LDIM * LDIM)

// ---------------------------------------------------------------------------
// Prep: convert weights (and squared weights for the variance path) to f16 in
// d_ws, and zero the two scalar accumulators in d_out. Runs every launch.
// ws layout (halves): [0]=W_in, [LL]=W_in^2, [2LL]=W1, [3LL]=W1^2, [4LL]=W2, [5LL]=W2^2
// ---------------------------------------------------------------------------
__global__ void interp_prep(const float* __restrict__ W_in,
                            const float* __restrict__ W1,
                            const float* __restrict__ W2,
                            _Float16* __restrict__ wsh,
                            float* __restrict__ scalars) {
  int idx = blockIdx.x * blockDim.x + threadIdx.x;
  if (idx == 0) { scalars[0] = 0.0f; scalars[1] = 0.0f; }
  if (idx < LL) {
    int n = idx / LDIM, k = idx % LDIM;
    float w0 = W_in[n * (LDIM + 3) + k];     // W_in is (L, L+3); first 128 cols via WMMA
    wsh[idx]          = (_Float16)w0;
    wsh[LL + idx]     = (_Float16)(w0 * w0);
    float w1 = W1[idx];
    wsh[2 * LL + idx] = (_Float16)w1;
    wsh[3 * LL + idx] = (_Float16)(w1 * w1);
    float w2 = W2[idx];
    wsh[4 * LL + idx] = (_Float16)w2;
    wsh[5 * LL + idx] = (_Float16)(w2 * w2);
  }
}

// A fragment: lane holds row M=l%16; K = kc*32 + 16h + [0..15] (contiguous)
__device__ __forceinline__ v16h frag_a(const _Float16* act, int M, int h, int kc) {
  v8h lo = *(const v8h*)(act + M * STRIDE + kc * 32 + 16 * h);
  v8h hi = *(const v8h*)(act + M * STRIDE + kc * 32 + 16 * h + 8);
  return __builtin_shufflevector(lo, hi, 0,1,2,3,4,5,6,7,8,9,10,11,12,13,14,15);
}

// One ADF layer: [mi,vi](16xL f16, LDS) @ W^T -> bias/tail -> relu-ADF or dropout -> [mo,vo]
// mode 0: W_in layer (adds pos_rel tail, then relu-ADF)
// mode 1: W1 layer   (relu-ADF)
// mode 2: W2 layer   (dropout + MIN_VAR)
__device__ __forceinline__ void run_layer(
    const _Float16* __restrict__ mi, const _Float16* __restrict__ vi,
    _Float16* __restrict__ mo, _Float16* __restrict__ vo,
    const _Float16* __restrict__ Wm, const _Float16* __restrict__ Wv,
    const float* __restrict__ bias, int mode,
    const float* __restrict__ W_in_raw, const float* __restrict__ sPos,
    const float* __restrict__ drop, int e0, int lane) {
  const int nl = lane & 15;   // A-row index AND B/D-column index for this lane
  const int h  = lane >> 4;
  for (int nt = 0; nt < 8; ++nt) {
    const int n = nt * 16 + nl;                 // output feature (weight row)
    v8f cm = {0.f,0.f,0.f,0.f,0.f,0.f,0.f,0.f};
    v8f cv = {0.f,0.f,0.f,0.f,0.f,0.f,0.f,0.f};
    const _Float16* wmrow = Wm + n * LDIM;
    const _Float16* wvrow = Wv + n * LDIM;
#pragma unroll
    for (int kc = 0; kc < 4; ++kc) {
      v16h am = frag_a(mi, nl, h, kc);
      v16h av = frag_a(vi, nl, h, kc);
      v16h bm = *(const v16h*)(wmrow + kc * 32 + 16 * h);   // B col = weight row, K-contig
      v16h bv = *(const v16h*)(wvrow + kc * 32 + 16 * h);
      cm = __builtin_amdgcn_wmma_f32_16x16x32_f16(false, am, false, bm, (short)0, cm, false, false);
      cv = __builtin_amdgcn_wmma_f32_16x16x32_f16(false, av, false, bv, (short)0, cv, false, false);
    }
    const float bn = bias[n];
    float w128 = 0.f, w129 = 0.f, w130 = 0.f, s128 = 0.f, s129 = 0.f, s130 = 0.f;
    if (mode == 0) {                            // pos_rel tail columns of W_in (K=128..130)
      w128 = W_in_raw[n * (LDIM + 3) + 128];
      w129 = W_in_raw[n * (LDIM + 3) + 129];
      w130 = W_in_raw[n * (LDIM + 3) + 130];
      s128 = w128 * w128; s129 = w129 * w129; s130 = w130 * w130;
    }
#pragma unroll
    for (int r8 = 0; r8 < 8; ++r8) {
      const int M = r8 + 8 * h;                 // edge-in-tile for this accumulator reg
      float m  = cm[r8] + bn;
      float vv = cv[r8];
      if (mode == 0) {
        float p0 = sPos[M * 3 + 0], p1 = sPos[M * 3 + 1], p2 = sPos[M * 3 + 2];
        m  += p0 * w128 + p1 * w129 + p2 * w130;
        vv += p0 * s128 + p1 * s129 + p2 * s130;   // v tail uses pos_rel * W^2
      }
      if (mode == 2) {                          // dropout + MIN_VAR
        float dm = drop[(size_t)(e0 + M) * LDIM + n];
        m  *= dm;
        vv  = vv * dm * dm + 0.001f;
      } else {                                  // ADF ReLU (f32)
        vv = fmaxf(vv, 1e-6f);
        float std_ = sqrtf(vv);
        float r    = m / std_;
        float cdf  = 0.5f * (1.0f + erff(r * 0.70710678118654752f));
        float pdf  = expf(-0.5f * r * r) * 0.39894228040143268f;
        float m2   = m * cdf + std_ * pdf;
        vv = (m * m + vv) * cdf + m * std_ * pdf - m2 * m2;
        m  = m2;
      }
      mo[M * STRIDE + n] = (_Float16)m;
      vo[M * STRIDE + n] = (_Float16)vv;
    }
  }
}

__global__ __launch_bounds__(32)
void interp_main(const float* __restrict__ pos_source,
                 const float* __restrict__ pos_target,
                 const float* __restrict__ latents,
                 const float* __restrict__ var,
                 const float* __restrict__ drop_mask,
                 const float* __restrict__ W_in, const float* __restrict__ b_in,
                 const float* __restrict__ b1,  const float* __restrict__ b2,
                 const float* __restrict__ W_out, const float* __restrict__ b_out,
                 const int* __restrict__ row, const int* __restrict__ col,
                 const int* __restrict__ occ,
                 const _Float16* __restrict__ wsh,
                 float* __restrict__ out) {
  __shared__ alignas(32) _Float16 buf[4][16 * STRIDE];  // ping-pong m/v tiles (f16)
  __shared__ float posrel[16 * 3];

  const int tile = blockIdx.x;
  const int lane = threadIdx.x;
  const int e0   = tile * 16;
  const int nl   = lane & 15;
  const int h    = lane >> 4;

  // ---- stage inputs: lane -> edge nl, k-half h (64 floats of latents & var) ----
  {
    const int e = e0 + nl;
    const int c = col[e];
    const float* lp = latents + (size_t)c * LDIM + h * 64;
    const float* vp = var     + (size_t)c * LDIM + h * 64;
#pragma unroll 4
    for (int k = 0; k < 64; k += 4) {
      float4 a = *(const float4*)(lp + k);
      float4 b = *(const float4*)(vp + k);
      int kk = h * 64 + k;
      buf[0][nl * STRIDE + kk + 0] = (_Float16)a.x;
      buf[0][nl * STRIDE + kk + 1] = (_Float16)a.y;
      buf[0][nl * STRIDE + kk + 2] = (_Float16)a.z;
      buf[0][nl * STRIDE + kk + 3] = (_Float16)a.w;
      buf[1][nl * STRIDE + kk + 0] = (_Float16)b.x;
      buf[1][nl * STRIDE + kk + 1] = (_Float16)b.y;
      buf[1][nl * STRIDE + kk + 2] = (_Float16)b.z;
      buf[1][nl * STRIDE + kk + 3] = (_Float16)b.w;
    }
    if (h == 0) {
      const int r = row[e];
#pragma unroll
      for (int j = 0; j < 3; ++j)
        posrel[nl * 3 + j] = pos_target[(size_t)r * 3 + j] - pos_source[(size_t)c * 3 + j];
    }
  }
  __syncthreads();

  // ---- 3 WMMA layers (m path uses W, v path uses W^2) ----
  run_layer(buf[0], buf[1], buf[2], buf[3], wsh,          wsh + LL,     b_in, 0,
            W_in, posrel, drop_mask, e0, lane);
  __syncthreads();
  run_layer(buf[2], buf[3], buf[0], buf[1], wsh + 2 * LL, wsh + 3 * LL, b1,   1,
            W_in, posrel, drop_mask, e0, lane);
  __syncthreads();
  run_layer(buf[0], buf[1], buf[2], buf[3], wsh + 4 * LL, wsh + 5 * LL, b2,   2,
            W_in, posrel, drop_mask, e0, lane);
  __syncthreads();

  // ---- head: only output channel 0 matters. lanes 0-15: m-dot, 16-31: v-dot ----
  {
    const int el = lane & 15;
    const _Float16* arow = (lane < 16 ? buf[2] : buf[3]) + el * STRIDE;
    const bool mpath = lane < 16;
    float acc = 0.0f;
#pragma unroll 8
    for (int k = 0; k < LDIM; ++k) {
      float w = W_out[k];                     // W_out row 0
      acc += (float)arow[k] * (mpath ? w : w * w);
    }
    float v4 = __shfl(acc, lane + 16, 32);    // lanes<16 fetch variance dot
    if (lane < 16) {
      const int e = e0 + el;
      float p = acc + b_out[0];
      out[e] = p;                             // preds
      float og = (float)occ[row[e]];
      out[ETOT + 2 + e] = og;                 // occ_gt
      float al = fabsf(v4);
      float rl = fmaxf(p, 0.0f) - p * og + log1pf(expf(-fabsf(p)));
#pragma unroll
      for (int s = 8; s >= 1; s >>= 1) {
        al += __shfl_xor(al, s, 32);
        rl += __shfl_xor(rl, s, 32);
      }
      if (el == 0) {
        atomicAdd(out + ETOT,     al * (1.0f / (float)ETOT));  // aleatoric mean
        atomicAdd(out + ETOT + 1, rl * (1.0f / (float)ETOT));  // recons_loss mean
      }
    }
  }
}

extern "C" void kernel_launch(void* const* d_in, const int* in_sizes, int n_in,
                              void* d_out, int out_size, void* d_ws, size_t ws_size,
                              hipStream_t stream) {
  const float* pos_source = (const float*)d_in[0];
  const float* pos_target = (const float*)d_in[1];
  const float* latents    = (const float*)d_in[2];
  const float* var        = (const float*)d_in[3];
  const float* drop_mask  = (const float*)d_in[4];
  const float* W_in       = (const float*)d_in[5];
  const float* b_in       = (const float*)d_in[6];
  const float* W1         = (const float*)d_in[7];
  const float* b1         = (const float*)d_in[8];
  const float* W2         = (const float*)d_in[9];
  const float* b2         = (const float*)d_in[10];
  const float* W_out      = (const float*)d_in[11];
  const float* b_out      = (const float*)d_in[12];
  const int*   row        = (const int*)d_in[13];
  const int*   col        = (const int*)d_in[14];
  const int*   occ        = (const int*)d_in[15];
  float*     out = (float*)d_out;
  _Float16*  wsh = (_Float16*)d_ws;

  interp_prep<<<(LL + 255) / 256, 256, 0, stream>>>(W_in, W1, W2, wsh, out + ETOT);
  interp_main<<<NTILES, 32, 0, stream>>>(pos_source, pos_target, latents, var, drop_mask,
                                         W_in, b_in, b1, b2, W_out, b_out,
                                         row, col, occ, wsh, out);
}
